// ImprovedZeroShotGNN_37314675868193
// MI455X (gfx1250) — matrix operand order repaired and checked by reading, not compiled, not run on hardware
//
#include <hip/hip_runtime.h>
#include <hip/hip_bf16.h>
#include <math.h>

// ============================================================================
// ImprovedZeroShotGNN forward for MI455X (gfx1250), wave32 + WMMA f16.
// All dense GEMMs use v_wmma_f32_16x16x32_f16 (f16 A/B, f32 acc).
// Weights pre-packed to fragment-major layout -> one 32B load per B frag.
// Each wave owns a 16-row strip and reuses A fragments across all N tiles.
// ============================================================================

typedef __attribute__((ext_vector_type(16))) _Float16 v16h;
typedef __attribute__((ext_vector_type(8)))  _Float16 v8h;
typedef __attribute__((ext_vector_type(8)))  float    v8f;
typedef _Float16 h16;

#define BB  1024
#define NU  20
#define NT  30
#define DD  64
#define HH  8
#define SS  50   // NU + NT

// ---------------------------------------------------------------------------
// param leaf indices (jax tree flatten: sorted dict keys, leaves b<g<w)
// ---------------------------------------------------------------------------
enum {
  IN_UF = 0, IN_TF, IN_UP, IN_TP,
  P_CAD_L1B = 4, P_CAD_L1W, P_CAD_L2B, P_CAD_L2W,
  P_CEST_L1B = 8, P_CEST_L1W, P_CEST_L2B, P_CEST_L2W, P_CEST_L3B, P_CEST_L3W,
  P_GAT0 = 14, // per layer (+10): +0 k.b +1 k.w +2 ln.b +3 ln.g +4 o.b +5 o.w +6 q.b +7 q.w +8 v.b +9 v.w
  P_SC_L1B = 44, P_SC_L1W, P_SC_L2B, P_SC_L2W, P_SC_L3B, P_SC_L3W,
  P_SP_A1B = 50, P_SP_A1W, P_SP_A2B, P_SP_A2W, P_SP_D1B, P_SP_D1W, P_SP_D2B, P_SP_D2W, P_SP_FB, P_SP_FW,
  P_TE_L1B = 60, P_TE_L1W, P_TE_L2B, P_TE_L2W, P_TE_L3B, P_TE_L3W, P_TE_LNB, P_TE_LNG,
  P_UE_L1B = 68, P_UE_L1W, P_UE_L2B, P_UE_L2W, P_UE_L3B, P_UE_L3W, P_UE_LNB, P_UE_LNG
};

// ---------------------------------------------------------------------------
// Weight packing: row-major f32 [K][N] -> fragment-major f16:
//   dst[ ((ktile*NTILES + nt)*32 + lane)*16 + e ]  where
//   krow = ktile*32 + (lane>>4)*8 + (e&7) + (e>>3)*16,  col = nt*16 + (lane&15)
// Each WMMA B-fragment becomes 32 contiguous bytes per lane.
// ---------------------------------------------------------------------------
__global__ void k_pack_b(const float* __restrict__ src, h16* __restrict__ dst,
                         int K, int N) {
  int idx = blockIdx.x * blockDim.x + threadIdx.x;
  if (idx >= K * N) return;
  int e = idx & 15;
  int L = (idx >> 4) & 31;
  int t = idx >> 9;                 // ktile*ntiles + nt
  int ntiles = N >> 4;
  int ktile = t / ntiles, nt = t % ntiles;
  int krow = ktile * 32 + (L >> 4) * 8 + (e & 7) + ((e >> 3) << 4);
  int col  = nt * 16 + (L & 15);
  dst[idx] = (h16)src[krow * N + col];
}

// same, but source has only Nsrc valid columns (zero-pad to Ndst)
__global__ void k_pack_b_pad(const float* __restrict__ src, h16* __restrict__ dst,
                             int K, int Nsrc, int Ndst) {
  int idx = blockIdx.x * blockDim.x + threadIdx.x;
  if (idx >= K * Ndst) return;
  int e = idx & 15;
  int L = (idx >> 4) & 31;
  int t = idx >> 9;
  int ntiles = Ndst >> 4;
  int ktile = t / ntiles, nt = t % ntiles;
  int krow = ktile * 32 + (L >> 4) * 8 + (e & 7) + ((e >> 3) << 4);
  int col  = nt * 16 + (L & 15);
  dst[idx] = (col < Nsrc) ? (h16)src[krow * Nsrc + col] : (h16)0.0f;
}

__global__ void k_vadd(const float* a, const float* b, float* c, int n) {
  int i = blockIdx.x * blockDim.x + threadIdx.x;
  if (i < n) c[i] = a[i] + b[i];
}

// ---------------------------------------------------------------------------
// A-fragment loaders per WMMA A layout (ISA 7.12.2):
// lane<16 holds K=ka..ka+7 and ka+16..ka+23 (ka = ktile*32 + (lane>=16)*8)
// ---------------------------------------------------------------------------
__device__ __forceinline__ v16h load_a_frag_f32(const float* __restrict__ base, int ka) {
  float4 a0 = *(const float4*)(base + ka);
  float4 a1 = *(const float4*)(base + ka + 4);
  float4 a2 = *(const float4*)(base + ka + 16);
  float4 a3 = *(const float4*)(base + ka + 20);
  v16h af;
  af[0]=(h16)a0.x; af[1]=(h16)a0.y; af[2]=(h16)a0.z; af[3]=(h16)a0.w;
  af[4]=(h16)a1.x; af[5]=(h16)a1.y; af[6]=(h16)a1.z; af[7]=(h16)a1.w;
  af[8]=(h16)a2.x; af[9]=(h16)a2.y; af[10]=(h16)a2.z; af[11]=(h16)a2.w;
  af[12]=(h16)a3.x; af[13]=(h16)a3.y; af[14]=(h16)a3.z; af[15]=(h16)a3.w;
  return af;
}

__device__ __forceinline__ v16h load_a_frag_h16(const h16* __restrict__ base, int ka) {
  v8h lo = *(const v8h*)(base + ka);
  v8h hi = *(const v8h*)(base + ka + 16);
  return __builtin_shufflevector(lo, hi, 0,1,2,3,4,5,6,7,8,9,10,11,12,13,14,15);
}

// ---------------------------------------------------------------------------
// Templated WMMA GEMM: C = act(A[M,K] @ W + bias).
// One wave computes a full 16xN strip: A fragments loaded once per K tile and
// reused across all NTILES B tiles (accumulators held in registers).
// Output row remap: orow = (r/RPB_IN)*RPB_OUT + rowOff + r%RPB_IN.
// ---------------------------------------------------------------------------
template<int N, int K, int RPB_IN, int RPB_OUT, int ACT>
__global__ __launch_bounds__(256)
void k_gemm_wmma(const float* __restrict__ A, const h16* __restrict__ Wpk,
                 const float* __restrict__ bias, float* __restrict__ C,
                 int M, int rowOff) {
  constexpr int NTILES = N / 16;
  constexpr int KTILES = K / 32;
  int wave = threadIdx.x >> 5;
  int lane = threadIdx.x & 31;
  int strip = blockIdx.x * (blockDim.x >> 5) + wave;
  if (strip >= (M >> 4)) return;                 // wave-uniform
  int r0 = strip << 4;
  int hs = lane >> 4;
  const float* Arow = A + (long)(r0 + (lane & 15)) * K;

  v8f acc[NTILES];
#pragma unroll
  for (int nt = 0; nt < NTILES; ++nt) acc[nt] = (v8f){};

#pragma unroll
  for (int kt = 0; kt < KTILES; ++kt) {
    v16h af = load_a_frag_f32(Arow, kt * 32 + hs * 8);
#pragma unroll
    for (int nt = 0; nt < NTILES; ++nt) {
      v16h bf = *(const v16h*)(Wpk + (((size_t)kt * NTILES + nt) * 32 + lane) * 16);
      acc[nt] = __builtin_amdgcn_wmma_f32_16x16x32_f16(false, af, false, bf,
                                                       (short)0, acc[nt], false, false);
    }
  }
#pragma unroll
  for (int nt = 0; nt < NTILES; ++nt) {
    int col = (nt << 4) + (lane & 15);
    float bv = bias[col];
#pragma unroll
    for (int i = 0; i < 8; ++i) {
      int r = r0 + hs * 8 + i;
      float v = acc[nt][i] + bv;
      if (ACT == 1) v = fmaxf(v, 0.0f);
      long orow = (long)(r / RPB_IN) * RPB_OUT + rowOff + (r % RPB_IN);
      C[orow * N + col] = v;
    }
  }
}

// ---------------------------------------------------------------------------
// Tiny-K dense layer (encoder l1 with K=8/4): one thread per output element.
// ---------------------------------------------------------------------------
template<int N, int K, int ACT>
__global__ void k_lin_small(const float* __restrict__ A, const float* __restrict__ W,
                            const float* __restrict__ bias, float* __restrict__ C,
                            long M) {
  long idx = (long)blockIdx.x * blockDim.x + threadIdx.x;
  if (idx >= M * N) return;
  long m = idx / N; int n = (int)(idx % N);
  float s = bias[n];
#pragma unroll
  for (int k = 0; k < K; ++k) s += A[m * K + k] * W[k * N + n];
  if (ACT) s = fmaxf(s, 0.0f);
  C[idx] = s;
}

// ---------------------------------------------------------------------------
// LayerNorm over last dim (64), row remap, in place.
// ---------------------------------------------------------------------------
template<int RPB_IN, int RPB_OUT>
__global__ void k_ln(float* __restrict__ buf, const float* __restrict__ g,
                     const float* __restrict__ b, long Mrows, int rowOff) {
  long r = (long)blockIdx.x * blockDim.x + threadIdx.x;
  if (r >= Mrows) return;
  long orow = (r / RPB_IN) * RPB_OUT + rowOff + (r % RPB_IN);
  float* p = buf + orow * DD;
  float m = 0.f;
  for (int d = 0; d < DD; ++d) m += p[d];
  m *= (1.0f / DD);
  float v = 0.f;
  for (int d = 0; d < DD; ++d) { float t = p[d] - m; v += t * t; }
  float inv = rsqrtf(v * (1.0f / DD) + 1e-5f);
  for (int d = 0; d < DD; ++d) p[d] = (p[d] - m) * inv * g[d] + b[d];
}

// x = LN(x + o) row-wise (residual + post-LN of GAT layer)
__global__ void k_addln(float* __restrict__ x, const float* __restrict__ o,
                        const float* __restrict__ g, const float* __restrict__ b,
                        long Mrows) {
  long r = (long)blockIdx.x * blockDim.x + threadIdx.x;
  if (r >= Mrows) return;
  float* p = x + r * DD;
  const float* q = o + r * DD;
  float m = 0.f;
  for (int d = 0; d < DD; ++d) m += p[d] + q[d];
  m *= (1.0f / DD);
  float v = 0.f;
  for (int d = 0; d < DD; ++d) { float t = p[d] + q[d] - m; v += t * t; }
  float inv = rsqrtf(v * (1.0f / DD) + 1e-5f);
  for (int d = 0; d < DD; ++d) p[d] = (p[d] + q[d] - m) * inv * g[d] + b[d];
}

// ---------------------------------------------------------------------------
// Spatial geometry + dist1/ang1 + mean over targets (mean commutes with the
// linear second layer).  Output: [B*NU][64] = [hd_mean | ha_mean].
// ---------------------------------------------------------------------------
__global__ void k_geom(const float* __restrict__ up, const float* __restrict__ tp,
                       const float* __restrict__ w1d, const float* __restrict__ b1d,
                       const float* __restrict__ w1a, const float* __restrict__ b1a,
                       float* __restrict__ sp_h) {
  int bu = blockIdx.x;            // b*NU + i
  int b = bu / NU, i = bu % NU;
  __shared__ float ds[32], as_[32];
  int lane = threadIdx.x;         // 32 threads
  if (lane < NT) {
    float dx = tp[((long)b * NT + lane) * 2 + 0] - up[((long)b * NU + i) * 2 + 0];
    float dy = tp[((long)b * NT + lane) * 2 + 1] - up[((long)b * NU + i) * 2 + 1];
    ds[lane]  = sqrtf(dx * dx + dy * dy);
    as_[lane] = atan2f(dy, dx);
  }
  __syncthreads();
  float wd = w1d[lane], bd = b1d[lane], wa = w1a[lane], ba = b1a[lane];
  float ad = 0.f, aa = 0.f;
  for (int t = 0; t < NT; ++t) {
    ad += fmaxf(wd * ds[t]  + bd, 0.0f);
    aa += fmaxf(wa * as_[t] + ba, 0.0f);
  }
  sp_h[(long)bu * 64 + lane]      = ad * (1.0f / NT);
  sp_h[(long)bu * 64 + 32 + lane] = aa * (1.0f / NT);
}

// fuse input: [u | sinfo | u*sinfo]  -> [B*NU][192]
__global__ void k_fusein(const float* __restrict__ u, const float* __restrict__ sinfo,
                         float* __restrict__ fin) {
  long idx = (long)blockIdx.x * blockDim.x + threadIdx.x;
  if (idx >= (long)BB * NU * DD) return;
  long r = idx / DD; int d = (int)(idx % DD);
  float uv = u[r * DD + d], sv = sinfo[r * DD + d];
  fin[r * 192 + d]       = uv;
  fin[r * 192 + 64 + d]  = sv;
  fin[r * 192 + 128 + d] = uv * sv;
}

// ---------------------------------------------------------------------------
// Multi-head attention core (S=50, HD=8): one block per (b,h).
// ---------------------------------------------------------------------------
__global__ void k_attn(const float* __restrict__ q, const float* __restrict__ k,
                       const float* __restrict__ v, float* __restrict__ out) {
  int b = blockIdx.x / HH, h = blockIdx.x % HH;
  __shared__ float Ks[SS][8], Vs[SS][8];
  int t = threadIdx.x;                     // 64 threads
  for (int idx = t; idx < SS * 8; idx += blockDim.x) {
    int row = idx / 8, d = idx % 8;
    long src = ((long)b * SS + row) * DD + h * 8 + d;
    Ks[row][d] = k[src];
    Vs[row][d] = v[src];
  }
  __syncthreads();
  if (t < SS) {
    float qv[8];
    long base = ((long)b * SS + t) * DD + h * 8;
    for (int d = 0; d < 8; ++d) qv[d] = q[base + d];
    float sc[SS], mx = -1e30f;
    for (int j = 0; j < SS; ++j) {
      float s = 0.f;
      for (int d = 0; d < 8; ++d) s += qv[d] * Ks[j][d];
      s *= 0.35355339059f;                 // 1/sqrt(8)
      sc[j] = s; mx = fmaxf(mx, s);
    }
    float sum = 0.f;
    for (int j = 0; j < SS; ++j) { sc[j] = __expf(sc[j] - mx); sum += sc[j]; }
    float inv = 1.0f / sum;
    float o[8] = {0,0,0,0,0,0,0,0};
    for (int j = 0; j < SS; ++j)
      for (int d = 0; d < 8; ++d) o[d] += sc[j] * Vs[j][d];
    for (int d = 0; d < 8; ++d) out[base + d] = o[d] * inv;
  }
}

// mean pooling -> g[B][128] = [mean_u x | mean_t x]
__global__ void k_pool(const float* __restrict__ x, float* __restrict__ g) {
  int idx = blockIdx.x * blockDim.x + threadIdx.x;
  if (idx >= BB * 128) return;
  int b = idx / 128, d = idx % 128;
  float s = 0.f;
  if (d < 64) { for (int i = 0; i < NU; ++i) s += x[((long)b * SS + i) * DD + d];      s *= (1.0f / NU); }
  else        { for (int j = 0; j < NT; ++j) s += x[((long)b * SS + NU + j) * DD + d - 64]; s *= (1.0f / NT); }
  g[idx] = s;
}

// cest l3 (32->1, sigmoid) + cadapt (1->32->1, sigmoid): one thread per batch
__global__ void k_tail(const float* __restrict__ c2,
                       const float* __restrict__ w3, const float* __restrict__ b3,
                       const float* __restrict__ wa1, const float* __restrict__ ba1,
                       const float* __restrict__ wa2, const float* __restrict__ ba2,
                       float* __restrict__ cw) {
  int b = blockIdx.x * blockDim.x + threadIdx.x;
  if (b >= BB) return;
  float s = b3[0];
  for (int k = 0; k < 32; ++k) s += c2[b * 32 + k] * w3[k];
  float cs = 1.0f / (1.0f + __expf(-s));
  float s2 = ba2[0];
  for (int c = 0; c < 32; ++c) s2 += fmaxf(cs * wa1[c] + ba1[c], 0.0f) * wa2[c];
  cw[b] = 1.0f / (1.0f + __expf(-s2));
}

// ---------------------------------------------------------------------------
// Fused pair scorer: (uf2[i]|tf2[j]) -> 128 -> 64 -> 4, all WMMA.
// One wave per 16-pair tile; h1/h2 staged in LDS as f16 in A-frag layout.
// ---------------------------------------------------------------------------
__global__ __launch_bounds__(32)
void k_scorer(const float* __restrict__ x,
              const h16* __restrict__ W1p, const float* __restrict__ b1,
              const h16* __restrict__ W2p, const float* __restrict__ b2,
              const h16* __restrict__ W3p, const float* __restrict__ b3,
              const float* __restrict__ cw, float* __restrict__ out) {
  __shared__ __align__(16) h16 h1s[16][128];
  __shared__ __align__(16) h16 h2s[16][64];
  int lane = threadIdx.x;                  // blockDim = 32 (one wave)
  int hs = lane >> 4;
  long r0 = (long)blockIdx.x * 16;
  long p  = r0 + (lane & 15);              // this lane's pair row for A frags
  int b   = (int)(p / (NU * NT));
  int rem = (int)(p % (NU * NT));
  int i = rem / NT, j = rem % NT;
  const float* pu = x + ((long)b * SS + i) * DD;        // uf2 row (k 0..63)
  const float* pt = x + ((long)b * SS + NU + j) * DD;   // tf2 row (k 64..127)
  int colL = lane & 15;

  // ---- layer 1: K=128, N=128 (A frags hoisted to registers by LICM) ----
  for (int nt = 0; nt < 8; ++nt) {
    v8f acc = {};
#pragma unroll
    for (int kq = 0; kq < 4; ++kq) {       // K tiles of 32
      const float* base = (kq < 2) ? (pu + kq * 32) : (pt + (kq - 2) * 32);
      v16h af = load_a_frag_f32(base, hs * 8);
      v16h bf = *(const v16h*)(W1p + (((size_t)kq * 8 + nt) * 32 + lane) * 16);
      acc = __builtin_amdgcn_wmma_f32_16x16x32_f16(false, af, false, bf, (short)0, acc, false, false);
    }
    float bv = b1[nt * 16 + colL];
#pragma unroll
    for (int e = 0; e < 8; ++e)
      h1s[hs * 8 + e][nt * 16 + colL] = (h16)fmaxf(acc[e] + bv, 0.0f);
  }
  __syncthreads();

  // ---- layer 2: K=128, N=64 (A from LDS h1s) ----
  int arow = lane & 15;
  for (int nt = 0; nt < 4; ++nt) {
    v8f acc = {};
#pragma unroll
    for (int kq = 0; kq < 4; ++kq) {
      v16h af = load_a_frag_h16(&h1s[arow][0], kq * 32 + hs * 8);
      v16h bf = *(const v16h*)(W2p + (((size_t)kq * 4 + nt) * 32 + lane) * 16);
      acc = __builtin_amdgcn_wmma_f32_16x16x32_f16(false, af, false, bf, (short)0, acc, false, false);
    }
    float bv = b2[nt * 16 + colL];
#pragma unroll
    for (int e = 0; e < 8; ++e)
      h2s[hs * 8 + e][nt * 16 + colL] = (h16)fmaxf(acc[e] + bv, 0.0f);
  }
  __syncthreads();

  // ---- layer 3: K=64, N=16 (4 valid cols) ----
  v8f acc = {};
#pragma unroll
  for (int kq = 0; kq < 2; ++kq) {
    v16h af = load_a_frag_h16(&h2s[arow][0], kq * 32 + hs * 8);
    v16h bf = *(const v16h*)(W3p + (((size_t)kq) * 32 + lane) * 16);
    acc = __builtin_amdgcn_wmma_f32_16x16x32_f16(false, af, false, bf, (short)0, acc, false, false);
  }
  if (colL < 4) {
    float bv = b3[colL];
#pragma unroll
    for (int e = 0; e < 8; ++e) {
      long pr = r0 + hs * 8 + e;
      out[pr * 4 + colL] = (acc[e] + bv) * cw[pr / (NU * NT)];
    }
  }
}

// ===========================================================================
// host
// ===========================================================================
extern "C" void kernel_launch(void* const* d_in, const int* in_sizes, int n_in,
                              void* d_out, int out_size, void* d_ws, size_t ws_size,
                              hipStream_t stream) {
  (void)in_sizes; (void)n_in; (void)out_size; (void)ws_size;
  auto FI = [&](int i) { return (const float*)d_in[i]; };

  // ---- workspace bump allocator (256B aligned) ----
  char* wsb = (char*)d_ws; size_t off = 0;
  auto alloc = [&](size_t bytes) { void* pp = wsb + off; off = (off + bytes + 255) & ~(size_t)255; return pp; };
  auto allocF = [&](size_t n) { return (float*)alloc(n * sizeof(float)); };
  auto allocH = [&](size_t n) { return (h16*)alloc(n * sizeof(h16)); };

  float* x      = allocF((size_t)BB * SS * DD);
  float* u_enc  = allocF((size_t)BB * NU * DD);
  float* bufA   = allocF((size_t)BB * NT * DD);
  float* bufB   = allocF((size_t)BB * NT * DD);
  float* sp_h   = allocF((size_t)BB * NU * DD);
  float* sinfo  = allocF((size_t)BB * NU * DD);
  float* fin    = allocF((size_t)BB * NU * 192);
  float* qb     = allocF((size_t)BB * SS * DD);
  float* kb     = allocF((size_t)BB * SS * DD);
  float* vb     = allocF((size_t)BB * SS * DD);
  float* attn   = allocF((size_t)BB * SS * DD);
  float* oout   = allocF((size_t)BB * SS * DD);
  float* gpool  = allocF((size_t)BB * 128);
  float* c1     = allocF((size_t)BB * 64);
  float* c2     = allocF((size_t)BB * 32);
  float* cwv    = allocF((size_t)BB);
  float* bias_sp= allocF(64);

  h16* w_uel2p = allocH(64 * 64);  h16* w_uel3p = allocH(64 * 64);
  h16* w_tel2p = allocH(64 * 64);  h16* w_tel3p = allocH(64 * 64);
  h16* w_sp2p  = allocH(64 * 64);                    // [dist2.w ; ang2.w] packed
  h16* w_fusep = allocH(192 * 64);
  h16* w_gat[3][4];                                  // q,k,v,o
  for (int l = 0; l < 3; ++l) for (int m = 0; m < 4; ++m) w_gat[l][m] = allocH(64 * 64);
  h16* w_ce1p = allocH(128 * 64);  h16* w_ce2p = allocH(64 * 32);
  h16* w_s1p  = allocH(128 * 128); h16* w_s2p = allocH(128 * 64); h16* w_s3p = allocH(64 * 16);

  auto pack = [&](int idx, h16* dst, int K, int N) {
    int n = K * N;
    k_pack_b<<<(n + 255) / 256, 256, 0, stream>>>(FI(idx), dst, K, N);
  };

  // ---- weight prep (pack to fragment-major f16) ----
  pack(P_UE_L2W, w_uel2p, 64, 64);  pack(P_UE_L3W, w_uel3p, 64, 64);
  pack(P_TE_L2W, w_tel2p, 64, 64);  pack(P_TE_L3W, w_tel3p, 64, 64);
  // combined spatial second layer: ktile0 = dist2.w (K rows 0..31),
  // ktile1 = ang2.w (K rows 32..63); each packs independently (32 rows = 1 ktile)
  pack(P_SP_D2W, w_sp2p,            32, 64);
  pack(P_SP_A2W, w_sp2p + 32 * 64,  32, 64);
  k_vadd<<<1, 64, 0, stream>>>(FI(P_SP_D2B), FI(P_SP_A2B), bias_sp, 64);
  pack(P_SP_FW, w_fusep, 192, 64);
  for (int l = 0; l < 3; ++l) {
    int g0 = P_GAT0 + l * 10;
    pack(g0 + 7, w_gat[l][0], 64, 64);  // q.w
    pack(g0 + 1, w_gat[l][1], 64, 64);  // k.w
    pack(g0 + 9, w_gat[l][2], 64, 64);  // v.w
    pack(g0 + 5, w_gat[l][3], 64, 64);  // o.w
  }
  pack(P_CEST_L1W, w_ce1p, 128, 64);  pack(P_CEST_L2W, w_ce2p, 64, 32);
  pack(P_SC_L1W, w_s1p, 128, 128);    pack(P_SC_L2W, w_s2p, 128, 64);
  k_pack_b_pad<<<(64 * 16 + 255) / 256, 256, 0, stream>>>(FI(P_SC_L3W), w_s3p, 64, 4, 16);

#define GEMM(Nv, Kv, RI, RO, ACT, A, W, BIAS, C, M, OFFV)                          \
  do {                                                                             \
    int strips_ = (M) / 16;                                                        \
    k_gemm_wmma<Nv, Kv, RI, RO, ACT><<<(strips_ + 7) / 8, 256, 0, stream>>>(       \
        A, W, BIAS, C, M, OFFV);                                                   \
  } while (0)

  // ---- UAV encoder -> u_enc ----
  {
    long M = (long)BB * NU;
    k_lin_small<64, 8, 1><<<(unsigned)((M * 64 + 255) / 256), 256, 0, stream>>>(FI(IN_UF), FI(P_UE_L1W), FI(P_UE_L1B), bufA, M);
    GEMM(64, 64, 1, 1, 1, bufA, w_uel2p, FI(P_UE_L2B), bufB, (int)M, 0);
    GEMM(64, 64, 1, 1, 0, bufB, w_uel3p, FI(P_UE_L3B), u_enc, (int)M, 0);
    k_ln<1, 1><<<(unsigned)((M + 63) / 64), 64, 0, stream>>>(u_enc, FI(P_UE_LNG), FI(P_UE_LNB), M, 0);
  }
  // ---- target encoder -> x rows [b*50+20 ..) ----
  {
    long M = (long)BB * NT;
    k_lin_small<64, 4, 1><<<(unsigned)((M * 64 + 255) / 256), 256, 0, stream>>>(FI(IN_TF), FI(P_TE_L1W), FI(P_TE_L1B), bufA, M);
    GEMM(64, 64, 1, 1, 1, bufA, w_tel2p, FI(P_TE_L2B), bufB, (int)M, 0);
    GEMM(64, 64, NT, SS, 0, bufB, w_tel3p, FI(P_TE_L3B), x, (int)M, NU);
    k_ln<NT, SS><<<(unsigned)((M + 63) / 64), 64, 0, stream>>>(x, FI(P_TE_LNG), FI(P_TE_LNB), M, NU);
  }
  // ---- spatial ----
  k_geom<<<BB * NU, 32, 0, stream>>>(FI(IN_UP), FI(IN_TP), FI(P_SP_D1W), FI(P_SP_D1B), FI(P_SP_A1W), FI(P_SP_A1B), sp_h);
  GEMM(64, 64, 1, 1, 0, sp_h, w_sp2p, bias_sp, sinfo, BB * NU, 0);
  k_fusein<<<(BB * NU * DD + 255) / 256, 256, 0, stream>>>(u_enc, sinfo, fin);
  GEMM(64, 192, NU, SS, 0, fin, w_fusep, FI(P_SP_FB), x, BB * NU, 0);

  // ---- 3 GAT layers ----
  long Mtok = (long)BB * SS;
  for (int l = 0; l < 3; ++l) {
    int g0 = P_GAT0 + l * 10;
    GEMM(64, 64, 1, 1, 0, x, w_gat[l][0], FI(g0 + 6), qb, (int)Mtok, 0);
    GEMM(64, 64, 1, 1, 0, x, w_gat[l][1], FI(g0 + 0), kb, (int)Mtok, 0);
    GEMM(64, 64, 1, 1, 0, x, w_gat[l][2], FI(g0 + 8), vb, (int)Mtok, 0);
    k_attn<<<BB * HH, 64, 0, stream>>>(qb, kb, vb, attn);
    GEMM(64, 64, 1, 1, 0, attn, w_gat[l][3], FI(g0 + 4), oout, (int)Mtok, 0);
    k_addln<<<(unsigned)((Mtok + 63) / 64), 64, 0, stream>>>(x, oout, FI(g0 + 3), FI(g0 + 2), Mtok);
  }

  // ---- coverage estimator + adapter ----
  k_pool<<<(BB * 128 + 255) / 256, 256, 0, stream>>>(x, gpool);
  GEMM(64, 128, 1, 1, 1, gpool, w_ce1p, FI(P_CEST_L1B), c1, BB, 0);
  GEMM(32, 64, 1, 1, 1, c1, w_ce2p, FI(P_CEST_L2B), c2, BB, 0);
  k_tail<<<(BB + 255) / 256, 256, 0, stream>>>(c2, FI(P_CEST_L3W), FI(P_CEST_L3B),
                                               FI(P_CAD_L1W), FI(P_CAD_L1B),
                                               FI(P_CAD_L2W), FI(P_CAD_L2B), cwv);

  // ---- fused pair scorer -> d_out ----
  int ptiles = (BB * NU * NT) / 16;        // 38400 wave-tiles
  k_scorer<<<ptiles, 32, 0, stream>>>(x, w_s1p, FI(P_SC_L1B), w_s2p, FI(P_SC_L2B),
                                      w_s3p, FI(P_SC_L3B), cwv, (float*)d_out);
#undef GEMM
}
